// RTN_79173427134913
// MI455X (gfx1250) — matrix operand order repaired and checked by reading, compiled, not run on hardware
//
#include <hip/hip_runtime.h>
#include <hip/hip_bf16.h>
#include <math.h>

// ---------------------------------------------------------------------------
// Problem constants
// ---------------------------------------------------------------------------
static const int Hc = 128;          // h_dim
static const int Gc = 128;          // graph_node_dim
static const int Tc = 128;          // window size
static const int Bc = 32;           // batch
static const int Pc = Tc * (Tc - 1) / 2;   // 8128 pairs
static const int NT = 64;           // 64 tiles of 128 rows cover P (last half-full)
#define EPSF 1e-6f

// Output layout (flat concat of tuple, fp32):
static const size_t O_HT  = 0;                       // (T,B,H) = 524288
static const size_t O_KLG = (size_t)Tc * Bc * Hc;    // 524288
static const size_t O_KLB = O_KLG + 1;               // 524289
static const size_t O_AT  = O_KLB + 1;               // alpha_tilde (B,P)
static const size_t O_AB  = O_AT + (size_t)Bc * Pc;  // alpha_bar   (B,P)

// ---------------------------------------------------------------------------
// bf16 helpers (bit-level, round-to-nearest-even)
// ---------------------------------------------------------------------------
typedef __attribute__((ext_vector_type(16))) __bf16 v16bf;
typedef __attribute__((ext_vector_type(8)))  float  v8f;

__device__ __forceinline__ unsigned short f2bf(float f) {
  unsigned int u = __builtin_bit_cast(unsigned int, f);
  unsigned int r = u + 0x7FFFu + ((u >> 16) & 1u);
  return (unsigned short)(r >> 16);
}
__device__ __forceinline__ float bf2f(unsigned short s) {
  unsigned int u = ((unsigned int)s) << 16;
  return __builtin_bit_cast(float, u);
}
__device__ __forceinline__ __bf16 bfbits(unsigned short s) {
  return __builtin_bit_cast(__bf16, s);
}
__device__ __forceinline__ float softplusf(float x) {
  if (x > 20.f)  return x;
  if (x < -20.f) return expf(x);
  return log1pf(expf(x));
}
__device__ __forceinline__ float sigmoidf(float x) { return 1.f / (1.f + expf(-x)); }

// ---------------------------------------------------------------------------
// Async global->LDS staging (CDNA5 GLOBAL_LOAD_ASYNC_TO_LDS_B128, ASYNCcnt).
// GV addressing: per-lane 64-bit global address; vdst = LDS byte address.
// Generic __shared__ pointers carry the LDS byte offset in their low 32 bits.
// ---------------------------------------------------------------------------
__device__ __forceinline__ void async_g2l_b128(unsigned int lds_byte_addr,
                                               const void* gaddr) {
  unsigned long long ga = (unsigned long long)gaddr;
  asm volatile("global_load_async_to_lds_b128 %0, %1, off"
               :: "v"(lds_byte_addr), "v"(ga)
               : "memory");
}
__device__ __forceinline__ void async_wait0() {
  asm volatile("s_wait_asynccnt 0" ::: "memory");
}

// ---------------------------------------------------------------------------
// WMMA fragment loaders (CDNA5 16x16x32 bf16 layouts, ISA 7.12.2)
// A tile in LDS: row-major [M][ld] bf16 bits.
// lanes 0-15: M=lane, VGPR0-3 -> K=0..7, VGPR4-7 -> K=16..23
// lanes16-31: M=lane-16, VGPR0-3 -> K=8..15, VGPR4-7 -> K=24..31
// ---------------------------------------------------------------------------
__device__ __forceinline__ v16bf load_fragA(const unsigned short* tile, int ld,
                                            int row0, int k0) {
  const int lane = threadIdx.x & 31;
  const unsigned short* p = tile + (row0 + (lane & 15)) * ld + k0 + ((lane & 16) ? 8 : 0);
  v16bf f;
#pragma unroll
  for (int j = 0; j < 8; ++j) {
    const int ko = (j < 4) ? (2 * j) : (2 * j + 8);
    f[2 * j]     = bfbits(p[ko]);
    f[2 * j + 1] = bfbits(p[ko + 1]);
  }
  return f;
}
// B tile in LDS: K-major [K][128] bf16 bits (weights pre-transposed in prep).
// N = n0 + lane%16; halves: lanes 0-15 K=0..7,16..23; lanes 16-31 K=8..15,24..31.
__device__ __forceinline__ v16bf load_fragB(const unsigned short* smWt,
                                            int n0, int k0) {
  const int lane = threadIdx.x & 31;
  const unsigned short* p = smWt + (k0 + ((lane & 16) ? 8 : 0)) * 128 + n0 + (lane & 15);
  v16bf f;
#pragma unroll
  for (int e = 0; e < 16; ++e) f[e] = bfbits(p[((e < 8) ? e : (e + 8)) * 128]);
  return f;
}

// ---------------------------------------------------------------------------
// One GEMM stage: D(128x128) = act( A(128xK) * W^T + bias ).
// Weights arrive pre-transposed ([K][128] bf16) and are staged to LDS with
// async b128 copies. 8 waves, each owns a 16-row strip (8x v8f accumulators).
// ---------------------------------------------------------------------------
template <int K, typename EPI>
__device__ __forceinline__ void gemm_compute(const unsigned short* smSrc,
                                             const unsigned short* smWt,
                                             const float* gBias, EPI epi) {
  const int tid  = threadIdx.x;
  const int wave = tid >> 5;
  const int lane = tid & 31;
  const int row0   = wave * 16;
  const int nlocal = lane & 15;
  const int mlocal = (lane >> 4) * 8;

  v8f acc[8];
#pragma unroll
  for (int n = 0; n < 8; ++n)
#pragma unroll
    for (int r = 0; r < 8; ++r) acc[n][r] = 0.f;

#pragma unroll
  for (int kk = 0; kk < K; kk += 32) {
    v16bf a = load_fragA(smSrc, K, row0, kk);
#pragma unroll
    for (int n = 0; n < 8; ++n) {
      v16bf b = load_fragB(smWt, n * 16, kk);
      acc[n] = __builtin_amdgcn_wmma_f32_16x16x32_bf16(
          false, a, false, b, (short)0, acc[n], false, false);
    }
  }
#pragma unroll
  for (int n = 0; n < 8; ++n) {
    const int col  = n * 16 + nlocal;
    const float bb = gBias[col];
#pragma unroll
    for (int r = 0; r < 8; ++r) {
      epi(row0 + mlocal + r, col, acc[n][r] + bb);
    }
  }
}

template <int K, typename EPI>
__device__ __forceinline__ void run_stage(const unsigned short* smSrc,
                                          const unsigned short* gWt,  // [K][128] bf16
                                          unsigned short* smWt,
                                          const float* gBias, EPI epi) {
  const int tid = threadIdx.x;
  // Contiguous async copy: weight layout in ws already matches LDS layout.
  const unsigned bytes   = (unsigned)(128 * K * 2);
  const unsigned ldsBase = (unsigned)(unsigned long long)(const void*)smWt;
  for (unsigned off = (unsigned)tid * 16u; off < bytes; off += 256u * 16u) {
    async_g2l_b128(ldsBase + off, (const char*)gWt + off);
  }
  async_wait0();
  __syncthreads();
  gemm_compute<K>(smSrc, smWt, gBias, epi);
  __syncthreads();
}

__device__ __forceinline__ void block_reduce_atomic(float* red, float v, float* gout) {
  const int tid = threadIdx.x;
  red[tid] = v;
  __syncthreads();
#pragma unroll
  for (int off = 128; off > 0; off >>= 1) {
    if (tid < off) red[tid] += red[tid + off];
    __syncthreads();
  }
  if (tid == 0) atomicAdd(gout, red[0]);
  __syncthreads();
}

// ---------------------------------------------------------------------------
// Prep: fp32 [N][K] -> bf16 [K][128] (K-major) weight conversion + transpose,
// and zeroing of global accumulators.
// ---------------------------------------------------------------------------
struct ConvArgs {
  const float* src[12];
  unsigned short* dst[12];
  int n[12];      // element count
  int kdim[12];   // K (inner) dimension of source [N][K]
};
__global__ void conv_bf16_kernel(ConvArgs A) {
  long idx = (long)blockIdx.x * blockDim.x + threadIdx.x;
  long off = 0;
#pragma unroll
  for (int s = 0; s < 12; ++s) {
    long e = off + A.n[s];
    if (idx < e) {
      long j = idx - off;
      int K = A.kdim[s];
      int nrow = (int)(j / K);
      int k    = (int)(j - (long)nrow * K);
      A.dst[s][(long)k * 128 + nrow] = f2bf(A.src[s][j]);
      return;
    }
    off = e;
  }
}
__global__ void zero_kernel(float* ei_num, float* ei_den, float* klg, float* klb) {
  int i = blockIdx.x * blockDim.x + threadIdx.x;
  if (i < Bc * Gc) ei_num[i] = 0.f;
  if (i < Bc)      ei_den[i] = 0.f;
  if (i == 0) { *klg = 0.f; *klb = 0.f; }
}

// ---------------------------------------------------------------------------
// Kernel 1: h_enc + node embedding (one 128-row tile per b), WMMA chain
// ---------------------------------------------------------------------------
struct NodeArgs {
  const float* x;                       // (T,B,H)
  const unsigned short *wFg, *wNe;      // bf16 [K][128]
  const float *bFg, *bNe;
  unsigned short* node;                 // out: (B,T,G) bf16
};
__global__ __launch_bounds__(256) void node_kernel(NodeArgs A) {
  extern __shared__ char smem[];
  unsigned short* smA = (unsigned short*)smem;       // 128x128
  unsigned short* smB = smA + 128 * 128;             // 128x128
  unsigned short* smW = smB + 128 * 128;             // 128x128 (K-major)
  const int b = blockIdx.x, tid = threadIdx.x;

  for (int i = tid; i < 128 * 128; i += 256) {
    int t = i >> 7, h = i & 127;
    smA[i] = f2bf(A.x[((size_t)t * Bc + b) * Hc + h]);
  }
  __syncthreads();

  const float* xg = A.x;
  unsigned short* smBp = smB;
  run_stage<128>(smA, A.wFg, smW, A.bFg, [=](int m, int n, float v) {
    float xv = xg[((size_t)m * Bc + b) * Hc + n];
    float fg = sigmoidf(v);
    smBp[m * 128 + n] = f2bf(fg * xv + (1.f - fg) * tanhf(xv));
  });
  unsigned short* nodep = A.node;
  run_stage<128>(smB, A.wNe, smW, A.bNe, [=](int m, int n, float v) {
    nodep[((size_t)b * Tc + m) * Gc + n] = f2bf(fmaxf(v, 0.f));
  });
}

// ---------------------------------------------------------------------------
// Kernel 2: the fused edge/VAE chain. Grid = B * NT blocks, 256 threads.
// ---------------------------------------------------------------------------
struct BigArgs {
  const unsigned short* node;            // (B,T,G) bf16
  const unsigned short *wE0, *wE1, *wE2; // edge MLP, bf16 [K][128]
  const unsigned short *wQe, *wQg, *wQb; // post encoder
  const unsigned short *wPe, *wPg, *wPb; // prior encoder
  const unsigned short *wPm;             // prior_mij
  const float *bE0, *bE1, *bE2, *bQe, *bQg, *bQb, *bPe, *bPg, *bPb, *bPm;
  const float *wQmean, *bQmean, *wQstd, *bQstd;   // post enc_mean/std (1,G)
  const float *wQma, *bQma, *wQsa, *bQsa;         // post_mean/std_approx
  const float *wPmean, *bPmean, *wPstd, *bPstd;   // prior enc_mean/std
  const float *eps1, *eps2;                       // (B,P)
  float *ei_num, *ei_den;                         // ws accumulators
  float *out_at, *out_ab, *out_klg, *out_klb;     // -> d_out slices
};

__global__ __launch_bounds__(256) void edge_fused_kernel(BigArgs A) {
  extern __shared__ char smem[];
  unsigned short* smA = (unsigned short*)smem;     // 128x256 (65536 B)
  unsigned short* smB = smA + 128 * 256;           // 128x128 (32768 B)
  unsigned short* smW = smB + 128 * 128;           // up to 256x128 (65536 B)
  unsigned short* smE = smW + 128 * 256;           // edge bf16 (32768 B)
  float* smEf  = (float*)(smE + 128 * 128);        // edge fp32 (65536 B)
  float* rowQm  = smEf + 128 * 128;
  float* rowQs  = rowQm + 128;
  float* rowPm  = rowQs + 128;
  float* rowPs  = rowPm + 128;
  float* rowPmj = rowPs + 128;                      // post_mij
  float* rowAt  = rowPmj + 128;
  float* rowAb  = rowAt + 128;
  float* red    = rowAb + 128;                      // 256 floats

  const int tid = threadIdx.x;
  const int b  = blockIdx.x >> 6;                  // / NT
  const int p0 = (blockIdx.x & 63) << 7;           // tile * 128

  // ---- build pair tile A0 (128 x 256): [node[one] | node[two]] ----
  {
    int row = tid >> 1, half = tid & 1;
    int p = p0 + row;
    unsigned short* dst = smA + row * 256 + half * 128;
    if (p < Pc) {
      int mm = (int)((sqrtf((float)(8 * p + 1)) + 1.0f) * 0.5f);
      while (mm * (mm - 1) / 2 > p) --mm;
      while (mm * (mm + 1) / 2 <= p) ++mm;
      int one = mm, two = p - mm * (mm - 1) / 2;
      int src = half ? two : one;
      const unsigned short* s = A.node + ((size_t)b * Tc + src) * Gc;
      for (int k = 0; k < 128; ++k) dst[k] = s[k];
    } else {
      for (int k = 0; k < 128; ++k) dst[k] = 0;
    }
  }
  __syncthreads();

  // ---- edge MLP ----
  run_stage<256>(smA, A.wE0, smW, A.bE0, [=](int m, int n, float v) {
    smB[m * 128 + n] = f2bf(fmaxf(v, 0.f));
  });
  run_stage<128>(smB, A.wE1, smW, A.bE1, [=](int m, int n, float v) {
    smA[m * 128 + n] = f2bf(fmaxf(v, 0.f));
  });
  run_stage<128>(smA, A.wE2, smW, A.bE2, [=](int m, int n, float v) {
    smE[m * 128 + n]  = f2bf(v);
    smEf[m * 128 + n] = v;
  });

  // ---- posterior branch ----
  run_stage<128>(smE, A.wQe, smW, A.bQe, [=](int m, int n, float v) {
    smB[m * 128 + n] = f2bf(fmaxf(v, 0.f));
  });
  run_stage<128>(smB, A.wQg, smW, A.bQg, [=](int m, int n, float v) {
    smA[m * 128 + n] = f2bf(fmaxf(v, 0.f));
  });
  if (tid < 128) {  // qm, qs from Qg
    float sm = 0.f, ss = 0.f;
    for (int k = 0; k < 128; ++k) {
      float v = bf2f(smA[tid * 128 + k]);
      sm += v * A.wQmean[k];
      ss += v * A.wQstd[k];
    }
    rowQm[tid] = sm + A.bQmean[0];
    rowQs[tid] = softplusf(ss + A.bQstd[0]);
  }
  __syncthreads();
  run_stage<128>(smB, A.wQb, smW, A.bQb, [=](int m, int n, float v) {
    smA[m * 128 + n] = f2bf(fmaxf(v, 0.f));
  });
  if (tid < 128) {  // qma, qsa from Qb; then per-row elementwise
    float sa = 0.f, sb = 0.f;
    for (int k = 0; k < 128; ++k) {
      float v = bf2f(smA[tid * 128 + k]);
      sa += v * A.wQma[k];
      sb += v * A.wQsa[k];
    }
    float qma = sa + A.bQma[0];
    float qsa = softplusf(sb + A.bQsa[0]);
    int p = p0 + tid;
    float at = 0.f, ab = 0.f, pmij = EPSF;
    if (p < Pc) {
      float nij = 2.f * qma - 1.f;
      pmij = 0.25f * (nij + sqrtf(nij * nij + 8.f * qsa * qsa)) + EPSF;
      float e1 = A.eps1[(size_t)b * Pc + p];
      float e2 = A.eps2[(size_t)b * Pc + p];
      at = softplusf(e1 * sqrtf(pmij) + pmij);
      float sij = e2 * sqrtf(at) * rowQs[tid] + at * rowQm[tid];
      ab = fmaxf(sij * at, 0.f);
      A.out_at[(size_t)b * Pc + p] = at;
      A.out_ab[(size_t)b * Pc + p] = ab;
    }
    rowPmj[tid] = pmij;
    rowAt[tid]  = at;
    rowAb[tid]  = ab;
  }
  __syncthreads();

  // ---- ei accumulation: sum_p alpha_bar * edge ----
  {
    int g = tid >> 1, half = tid & 1;
    float s = 0.f;
    for (int r = half * 64; r < half * 64 + 64; ++r)
      s += rowAb[r] * smEf[r * 128 + g];
    atomicAdd(&A.ei_num[b * Gc + g], s);
  }
  block_reduce_atomic(red, (tid < 128) ? rowAb[tid] : 0.f, &A.ei_den[b]);

  // ---- prior branch ----
  run_stage<128>(smE, A.wPe, smW, A.bPe, [=](int m, int n, float v) {
    smB[m * 128 + n] = f2bf(fmaxf(v, 0.f));
  });
  run_stage<128>(smB, A.wPg, smW, A.bPg, [=](int m, int n, float v) {
    smA[m * 128 + n] = f2bf(fmaxf(v, 0.f));
  });
  if (tid < 128) {  // pm, ps from Pg
    float sm = 0.f, ss = 0.f;
    for (int k = 0; k < 128; ++k) {
      float v = bf2f(smA[tid * 128 + k]);
      sm += v * A.wPmean[k];
      ss += v * A.wPstd[k];
    }
    rowPm[tid] = sm + A.bPmean[0];
    rowPs[tid] = softplusf(ss + A.bPstd[0]);
  }
  __syncthreads();
  // kl_g (Gaussian KL, per pair row)
  {
    float v = 0.f;
    if (tid < 128) {
      int p = p0 + tid;
      if (p < Pc) {
        float at = rowAt[tid], sq = sqrtf(at);
        float mp  = at * rowQm[tid], sp  = sq * rowQs[tid];
        float mpr = at * rowPm[tid], spr = sq * rowPs[tid];
        float d = 2.f * logf(spr + EPSF) - 2.f * logf(sp + EPSF)
                + (sp * sp + (mp - mpr) * (mp - mpr)) / ((spr + EPSF) * (spr + EPSF))
                - 1.f;
        v = 0.5f * fabsf(d);
      }
    }
    block_reduce_atomic(red, v, A.out_klg);
  }
  run_stage<128>(smB, A.wPb, smW, A.bPb, [=](int m, int n, float v) {
    smA[m * 128 + n] = f2bf(fmaxf(v, 0.f));
  });
  // prior_mij GEMM fused with kl_b reduction (no store of the tile)
  {
    float klbLocal = 0.f;
    run_stage<128>(smA, A.wPm, smW, A.bPm, [&](int m, int n, float v) {
      (void)n;
      int p = p0 + m;
      if (p < Pc) {
        float pr = 0.4f * sigmoidf(v);
        float po = rowPmj[m];
        float kb = pr - po + po * (logf(po + EPSF) - logf(pr + EPSF));
        klbLocal += fabsf(kb);
      }
    });
    block_reduce_atomic(red, klbLocal, A.out_klb);
  }
}

// ---------------------------------------------------------------------------
// Kernel 3: ei finalize + transform -> tei (B,G)
// ---------------------------------------------------------------------------
__global__ void tei_kernel(const float* ei_num, const float* ei_den,
                           const float* wT, const float* bT, float* tei) {
  __shared__ float lei[128];
  const int b = blockIdx.x, g = threadIdx.x;
  lei[g] = ei_num[b * 128 + g] / (ei_den[b] + EPSF);
  __syncthreads();
  float s = bT[g];
  for (int k = 0; k < 128; ++k) s += lei[k] * wT[g * 128 + k];
  tei[b * 128 + g] = s;
}

// ---------------------------------------------------------------------------
// Kernel 4: SRU layer 0 (input is t-invariant -> u computed once per b)
// ---------------------------------------------------------------------------
__global__ void sru0_kernel(const float* tei, const float* W0,
                            const float* bf0, const float* br0, float* h0) {
  __shared__ float lt[128];
  __shared__ float u0[384];
  const int b = blockIdx.x, tid = threadIdx.x;   // 128 threads
  lt[tid] = tei[b * 128 + tid];
  __syncthreads();
  for (int j = 0; j < 3; ++j) {
    int k = tid + j * 128;
    float s = 0.f;
    for (int d = 0; d < 128; ++d) s += lt[d] * W0[d * 384 + k];
    u0[k] = s;
  }
  __syncthreads();
  float xt = u0[tid];
  float f  = sigmoidf(u0[128 + tid] + bf0[tid]);
  float r  = sigmoidf(u0[256 + tid] + br0[tid]);
  float tv = lt[tid];
  float c = 0.f;
  for (int t = 0; t < Tc; ++t) {
    c = f * c + (1.f - f) * xt;
    h0[((size_t)t * Bc + b) * Hc + tid] = r * tanhf(c) + (1.f - r) * tv;
  }
}

// ---------------------------------------------------------------------------
// Kernel 5: SRU layer 1 pre-activation u1 = h0 @ W1 (T*B x 128 x 384)
// ---------------------------------------------------------------------------
__global__ void sru1_u_kernel(const float* h0, const float* W1, float* u1) {
  long idx = (long)blockIdx.x * 256 + threadIdx.x;
  if (idx >= (long)Tc * Bc * 384) return;
  int k = (int)(idx % 384);
  long rb = idx / 384;                 // t*B + b
  const float* hrow = h0 + rb * Hc;
  float s = 0.f;
  for (int d = 0; d < 128; ++d) s += hrow[d] * W1[d * 384 + k];
  u1[idx] = s;
}

// ---------------------------------------------------------------------------
// Kernel 6: SRU layer 1 scan + final output
// ---------------------------------------------------------------------------
__global__ void sru1_scan_kernel(const float* u1, const float* h0,
                                 const float* bf1, const float* br1, float* out) {
  int idx = blockIdx.x * 256 + threadIdx.x;   // B*H = 4096
  if (idx >= Bc * Hc) return;
  int b = idx >> 7, h = idx & 127;
  float c = 0.f;
  for (int t = 0; t < Tc; ++t) {
    long rb = (long)t * Bc + b;
    const float* u = u1 + rb * 384;
    float xt = u[h];
    float f  = sigmoidf(u[128 + h] + bf1[h]);
    float r  = sigmoidf(u[256 + h] + br1[h]);
    c = f * c + (1.f - f) * xt;
    out[rb * 128 + h] = r * tanhf(c) + (1.f - r) * h0[rb * 128 + h];
  }
}

// ---------------------------------------------------------------------------
// Input index map (setup_inputs dict order; params flattened as jax pytree:
// dict keys sorted, tuples in order)
// ---------------------------------------------------------------------------
enum {
  IN_X = 0,
  IN_E0_W, IN_E0_B, IN_E1_W, IN_E1_B, IN_E2_W, IN_E2_B,
  IN_FG_W, IN_FG_B, IN_NE_W, IN_NE_B,
  IN_Q_ENC_W, IN_Q_ENC_B, IN_Q_ENCB_W, IN_Q_ENCB_B, IN_Q_ENCG_W, IN_Q_ENCG_B,
  IN_Q_MEAN_W, IN_Q_MEAN_B, IN_Q_STD_W, IN_Q_STD_B,
  IN_Q_MA_W, IN_Q_MA_B, IN_Q_SA_W, IN_Q_SA_B,
  IN_P_ENC_W, IN_P_ENC_B, IN_P_ENCB_W, IN_P_ENCB_B, IN_P_ENCG_W, IN_P_ENCG_B,
  IN_P_MEAN_W, IN_P_MEAN_B, IN_P_STD_W, IN_P_STD_B,
  IN_PMIJ_W, IN_PMIJ_B,
  IN_SRU0_W, IN_SRU0_BF, IN_SRU0_BR,
  IN_SRU1_W, IN_SRU1_BF, IN_SRU1_BR,
  IN_TRANS_W, IN_TRANS_B,
  IN_EPS1, IN_EPS2
};

extern "C" void kernel_launch(void* const* d_in, const int* in_sizes, int n_in,
                              void* d_out, int out_size, void* d_ws, size_t ws_size,
                              hipStream_t stream) {
  (void)in_sizes; (void)n_in; (void)out_size; (void)ws_size;
#define FP(i) ((const float*)d_in[(i)])

  float* out = (float*)d_out;
  char* ws = (char*)d_ws;

  // ---- workspace carve-out ----
  size_t off = 0;
  auto carve = [&](size_t bytes) -> char* {
    char* p = ws + off;
    off = (off + bytes + 255) & ~(size_t)255;
    return p;
  };
  unsigned short* nodeBf = (unsigned short*)carve((size_t)Bc * Tc * Gc * 2);
  unsigned short* wE0 = (unsigned short*)carve(128 * 256 * 2);
  unsigned short* wE1 = (unsigned short*)carve(128 * 128 * 2);
  unsigned short* wE2 = (unsigned short*)carve(128 * 128 * 2);
  unsigned short* wQe = (unsigned short*)carve(128 * 128 * 2);
  unsigned short* wQg = (unsigned short*)carve(128 * 128 * 2);
  unsigned short* wQb = (unsigned short*)carve(128 * 128 * 2);
  unsigned short* wPe = (unsigned short*)carve(128 * 128 * 2);
  unsigned short* wPg = (unsigned short*)carve(128 * 128 * 2);
  unsigned short* wPb = (unsigned short*)carve(128 * 128 * 2);
  unsigned short* wPm = (unsigned short*)carve(128 * 128 * 2);
  unsigned short* wFg = (unsigned short*)carve(128 * 128 * 2);
  unsigned short* wNe = (unsigned short*)carve(128 * 128 * 2);
  float* ei_num = (float*)carve((size_t)Bc * Gc * 4);
  float* ei_den = (float*)carve((size_t)Bc * 4);
  float* teiW   = (float*)carve((size_t)Bc * Gc * 4);
  float* h0     = (float*)carve((size_t)Tc * Bc * Hc * 4);
  float* u1     = (float*)carve((size_t)Tc * Bc * 384 * 4);

  // ---- prep: zero accumulators, convert+transpose weights to bf16 K-major ----
  zero_kernel<<<(Bc * Gc + 255) / 256, 256, 0, stream>>>(
      ei_num, ei_den, out + O_KLG, out + O_KLB);

  ConvArgs ca;
  const float* srcs[12] = {FP(IN_E0_W), FP(IN_E1_W), FP(IN_E2_W),
                           FP(IN_Q_ENC_W), FP(IN_Q_ENCG_W), FP(IN_Q_ENCB_W),
                           FP(IN_P_ENC_W), FP(IN_P_ENCG_W), FP(IN_P_ENCB_W),
                           FP(IN_PMIJ_W), FP(IN_FG_W), FP(IN_NE_W)};
  unsigned short* dsts[12] = {wE0, wE1, wE2, wQe, wQg, wQb, wPe, wPg, wPb, wPm, wFg, wNe};
  int cnts[12] = {128 * 256, 128 * 128, 128 * 128, 128 * 128, 128 * 128, 128 * 128,
                  128 * 128, 128 * 128, 128 * 128, 128 * 128, 128 * 128, 128 * 128};
  int kdims[12] = {256, 128, 128, 128, 128, 128, 128, 128, 128, 128, 128, 128};
  long total = 0;
  for (int s = 0; s < 12; ++s) {
    ca.src[s] = srcs[s]; ca.dst[s] = dsts[s]; ca.n[s] = cnts[s]; ca.kdim[s] = kdims[s];
    total += cnts[s];
  }
  conv_bf16_kernel<<<(int)((total + 255) / 256), 256, 0, stream>>>(ca);

  // ---- node embedding ----
  NodeArgs na;
  na.x = FP(IN_X);
  na.wFg = wFg; na.wNe = wNe;
  na.bFg = FP(IN_FG_B); na.bNe = FP(IN_NE_B);
  na.node = nodeBf;
  node_kernel<<<Bc, 256, 3 * 128 * 128 * 2, stream>>>(na);

  // ---- fused edge/VAE chain ----
  BigArgs ba;
  ba.node = nodeBf;
  ba.wE0 = wE0; ba.wE1 = wE1; ba.wE2 = wE2;
  ba.wQe = wQe; ba.wQg = wQg; ba.wQb = wQb;
  ba.wPe = wPe; ba.wPg = wPg; ba.wPb = wPb; ba.wPm = wPm;
  ba.bE0 = FP(IN_E0_B); ba.bE1 = FP(IN_E1_B); ba.bE2 = FP(IN_E2_B);
  ba.bQe = FP(IN_Q_ENC_B); ba.bQg = FP(IN_Q_ENCG_B); ba.bQb = FP(IN_Q_ENCB_B);
  ba.bPe = FP(IN_P_ENC_B); ba.bPg = FP(IN_P_ENCG_B); ba.bPb = FP(IN_P_ENCB_B);
  ba.bPm = FP(IN_PMIJ_B);
  ba.wQmean = FP(IN_Q_MEAN_W); ba.bQmean = FP(IN_Q_MEAN_B);
  ba.wQstd  = FP(IN_Q_STD_W);  ba.bQstd  = FP(IN_Q_STD_B);
  ba.wQma   = FP(IN_Q_MA_W);   ba.bQma   = FP(IN_Q_MA_B);
  ba.wQsa   = FP(IN_Q_SA_W);   ba.bQsa   = FP(IN_Q_SA_B);
  ba.wPmean = FP(IN_P_MEAN_W); ba.bPmean = FP(IN_P_MEAN_B);
  ba.wPstd  = FP(IN_P_STD_W);  ba.bPstd  = FP(IN_P_STD_B);
  ba.eps1 = FP(IN_EPS1); ba.eps2 = FP(IN_EPS2);
  ba.ei_num = ei_num; ba.ei_den = ei_den;
  ba.out_at = out + O_AT; ba.out_ab = out + O_AB;
  ba.out_klg = out + O_KLG; ba.out_klb = out + O_KLB;
  // LDS: A(64K) + B(32K) + W(64K) + E(32K) + Ef(64K) + 7x128 f32 + 256 f32
  const size_t bigLds = (size_t)(128 * 256 + 128 * 128 + 128 * 256 + 128 * 128) * 2
                      + (size_t)(128 * 128 + 7 * 128 + 256) * 4;
  edge_fused_kernel<<<Bc * NT, 256, bigLds, stream>>>(ba);

  // ---- tails: ei -> tei -> SRU -> ht_res ----
  tei_kernel<<<Bc, 128, 0, stream>>>(ei_num, ei_den, FP(IN_TRANS_W), FP(IN_TRANS_B), teiW);
  sru0_kernel<<<Bc, 128, 0, stream>>>(teiW, FP(IN_SRU0_W), FP(IN_SRU0_BF), FP(IN_SRU0_BR), h0);
  sru1_u_kernel<<<(int)(((long)Tc * Bc * 384 + 255) / 256), 256, 0, stream>>>(h0, FP(IN_SRU1_W), u1);
  sru1_scan_kernel<<<(Bc * Hc + 255) / 256, 256, 0, stream>>>(
      u1, h0, FP(IN_SRU1_BF), FP(IN_SRU1_BR), out + O_HT);
#undef FP
}
